// ReVeal_74706661147278
// MI455X (gfx1250) — matrix operand order repaired and checked by reading, not compiled, not run on hardware
//
#include <hip/hip_runtime.h>

// ---------------------------------------------------------------------------
// ReVeal GGNN forward for MI455X (gfx1250, wave32, WMMA).
// All dense math runs through one generic WMMA GEMM kernel:
//   C[M,NP] = relu?( A[M,KP](f16) * Bt[KP,NP](f16, fragment-packed) + bias )
// with f32 accumulation via v_wmma_f32_16x16x32_f16.
// ---------------------------------------------------------------------------

typedef __attribute__((ext_vector_type(16))) _Float16 v16h;
typedef __attribute__((ext_vector_type(8)))  _Float16 v8h;
typedef __attribute__((ext_vector_type(8)))  float    v8f;

#define N_NODES 102400
#define N_EDGES 819200
#define DIN     120
#define HID     200
#define HS      208      // f32 column stride for H-sized arrays
#define HK      224      // f16 K-pad for H (multiple of 32)
#define CCK     320      // concat dim, already multiple of 32/16
#define G3P     608      // 3*H=600 padded to multiple of 16 (cols) / holds 600 valid
#define NB      256      // graphs
#define LSEQ    400      // nodes per graph
#define L1OUT   398      // after conv K=3
#define P1OUT   198      // after pool(3,2)
#define P2OUT   99       // after pool(2,2)
#define ROWS1   (NB*L1OUT)   // 101888
#define ROWSP1  (NB*P1OUT)   // 50688
#define ROWS2   (NB*P2OUT)   // 25344

// ---------------------------------------------------------------------------
// Generic WMMA GEMM.  Grid: x = M/16 row tiles, y = ceil((NP/16)/4).
// Block: 128 threads = 4 waves; wave w computes N-tile blockIdx.y*4+w.
// A tile (16 x KP f16) is staged in LDS and shared by the 4 waves.
// B is pre-packed in WMMA B-fragment order: each lane reads 16 contiguous
// halves (32B) per 32-wide K chunk.
// ---------------------------------------------------------------------------
__global__ void __launch_bounds__(128)
gemm_f16_wmma(const _Float16* __restrict__ A,
              const _Float16* __restrict__ Bp,
              const float*    __restrict__ bias,
              float*          __restrict__ C,
              int M, int KP, int NP, int relu)
{
    __shared__ _Float16 As[16 * 960];            // KP <= 960
    const int lane  = threadIdx.x & 31;
    const int wave  = threadIdx.x >> 5;
    const long long mtile = blockIdx.x;
    const int ntile = blockIdx.y * 4 + wave;
    const int ntiles = NP >> 4;

    // Cooperative A-tile load (16 rows x KP halves) as 32-bit words.
    {
        const unsigned int* Ag = (const unsigned int*)(A + (size_t)mtile * 16 * KP);
        unsigned int* As32 = (unsigned int*)As;
        const int words = (16 * KP) >> 1;
        for (int i = threadIdx.x; i < words; i += 128) As32[i] = Ag[i];
    }
    __syncthreads();
    if (ntile >= ntiles) return;                 // whole-wave uniform exit

    v8f acc = {0.f, 0.f, 0.f, 0.f, 0.f, 0.f, 0.f, 0.f};
    const int m  = lane & 15;
    const int kb = (lane < 16) ? 0 : 8;          // A-fragment K base (ISA layout)
    const int kchunks = KP >> 5;
    const _Float16* Bw = Bp + ((size_t)ntile * kchunks) * 512 + lane * 16;

    for (int kk = 0; kk < kchunks; ++kk) {
        // A 16-bit 16x32 fragment: lanes 0-15 row m hold K {kb..kb+7, kb+16..kb+23}
        v8h alo = *(const v8h*)(&As[m * KP + (kk << 5) + kb]);
        v8h ahi = *(const v8h*)(&As[m * KP + (kk << 5) + kb + 16]);
        v16h af;
        #pragma unroll
        for (int j = 0; j < 8; ++j) { af[j] = alo[j]; af[j + 8] = ahi[j]; }
        // B fragment: contiguous 32B per lane (pre-swizzled by pack_bt)
        v16h bf = *(const v16h*)(Bw + (size_t)kk * 512);
        __builtin_prefetch(Bw + (size_t)(kk + 1) * 512, 0, 1);
        acc = __builtin_amdgcn_wmma_f32_16x16x32_f16(false, af, false, bf,
                                                     (short)0, acc, false, false);
    }

    // C/D layout: VGPR r -> M = r (+8 for lanes 16-31), N = lane&15.
    const int ncol = (ntile << 4) + (lane & 15);
    const float bv = bias ? bias[ncol] : 0.0f;
    const long long mbase = mtile * 16 + ((lane < 16) ? 0 : 8);
    #pragma unroll
    for (int r = 0; r < 8; ++r) {
        float v = acc[r] + bv;
        if (relu) v = fmaxf(v, 0.0f);
        C[(size_t)(mbase + r) * NP + ncol] = v;
    }
}

// Pack W[O, Ieff] (f32, row-major) -> B-fragment-ordered f16:
// Bp[((ntile_base+tl)*(KP/32)+kk)*512 + lane*16 + j] = W[n][k],
//   n = tl*16 + (lane&15),  k = kk*32 + (lane<16?0:8) + (j<8 ? j : j+8).
__global__ void pack_bt(const float* __restrict__ W, _Float16* __restrict__ Bp,
                        int O, int Ieff, int KP, int ntile_base, int ntiles_local)
{
    long long t = (long long)blockIdx.x * blockDim.x + threadIdx.x;
    long long total = (long long)ntiles_local * (KP >> 5) * 512;
    if (t >= total) return;
    int j    = (int)(t & 15);
    int lane = (int)((t >> 4) & 31);
    long long rest = t >> 9;
    int kk = (int)(rest % (KP >> 5));
    int tl = (int)(rest / (KP >> 5));
    int n  = (tl << 4) + (lane & 15);
    int k  = (kk << 5) + ((lane < 16) ? 0 : 8) + ((j < 8) ? j : j + 8);
    float v = (n < O && k < Ieff) ? W[(size_t)n * Ieff + k] : 0.0f;
    Bp[(((size_t)(ntile_base + tl) * (KP >> 5) + kk) << 9) + lane * 16 + j] = (_Float16)v;
}

__global__ void pack_bias(const float* __restrict__ b, float* __restrict__ bp,
                          int O, int n_base, int count)
{
    int t = blockIdx.x * blockDim.x + threadIdx.x;
    if (t >= count) return;
    bp[n_base + t] = (t < O) ? b[t] : 0.0f;
}

// h0 = zero-pad(x) to H; writes both f32 (stride 208) and f16 (stride 224).
__global__ void init_h(const float* __restrict__ x, float* __restrict__ h32,
                       _Float16* __restrict__ h16, long long total)
{
    long long t = (long long)blockIdx.x * blockDim.x + threadIdx.x;
    if (t >= total) return;
    int c = (int)(t % HK);
    long long n = t / HK;
    float v = (c < DIN) ? x[n * DIN + c] : 0.0f;
    h16[t] = (_Float16)v;
    if (c < HS) h32[n * HS + c] = v;
}

__global__ void zero_f32(float* __restrict__ p, long long total)
{
    long long t = (long long)blockIdx.x * blockDim.x + threadIdx.x;
    if (t < total) p[t] = 0.0f;
}

// a[dst] += Wh_{etype(e)}(src); Whcat is [N, 416]: cols [0,208) etype0, [208,416) etype1.
__global__ void edge_scatter(const float* __restrict__ Wh, const int* __restrict__ srcv,
                             const int* __restrict__ dstv, const int* __restrict__ et,
                             float* __restrict__ a, long long total)
{
    long long t = (long long)blockIdx.x * blockDim.x + threadIdx.x;
    if (t >= total) return;
    int f = (int)(t % HID);
    long long ei = t / HID;
    int e = et[ei];
    long long s = srcv[ei];
    long long d = dstv[ei];
    float v = Wh[s * 416 + (long long)e * HS + f];
    __hip_atomic_fetch_add(&a[d * HS + f], v, __ATOMIC_RELAXED, __HIP_MEMORY_SCOPE_AGENT);
}

__global__ void cast_pad_f16(const float* __restrict__ in, _Float16* __restrict__ out,
                             int inStride, int valid, int outStride, long long total)
{
    long long t = (long long)blockIdx.x * blockDim.x + threadIdx.x;
    if (t >= total) return;
    int c = (int)(t % outStride);
    long long r = t / outStride;
    out[t] = (c < valid) ? (_Float16)in[r * inStride + c] : (_Float16)0.0f;
}

__device__ __forceinline__ float sigf(float x) { return 1.0f / (1.0f + expf(-x)); }

// GRUCell update; gi/gh are [N, 608] (valid 600), h32 [N,208], h16 [N,224].
__global__ void gru_update(const float* __restrict__ gi, const float* __restrict__ gh,
                           float* __restrict__ h32, _Float16* __restrict__ h16,
                           long long total)
{
    long long t = (long long)blockIdx.x * blockDim.x + threadIdx.x;
    if (t >= total) return;
    int f = (int)(t % HID);
    long long n = t / HID;
    const float* gin = gi + n * G3P;
    const float* ghn = gh + n * G3P;
    float r  = sigf(gin[f]         + ghn[f]);
    float z  = sigf(gin[HID + f]   + ghn[HID + f]);
    float nn = tanhf(gin[2*HID + f] + r * ghn[2*HID + f]);
    float hv = h32[n * HS + f];
    float o  = (1.0f - z) * nn + z * hv;
    h32[n * HS + f] = o;
    h16[n * HK + f] = (_Float16)o;
}

// ci = concat(x, h) as f16 [N, 320]
__global__ void ci_pack(const float* __restrict__ x, const float* __restrict__ h32,
                        _Float16* __restrict__ ci, long long total)
{
    long long t = (long long)blockIdx.x * blockDim.x + threadIdx.x;
    if (t >= total) return;
    int c = (int)(t % CCK);
    long long n = t / CCK;
    float v = (c < DIN) ? x[n * DIN + c] : h32[n * HS + (c - DIN)];
    ci[t] = (_Float16)v;
}

// im2col for K=3 VALID conv: row (b,l), col j=c*3+dk (matches torch OIH flatten)
__global__ void im2col3(const _Float16* __restrict__ src, _Float16* __restrict__ out,
                        int srcStride, int C, int KP, long long total)
{
    long long t = (long long)blockIdx.x * blockDim.x + threadIdx.x;
    if (t >= total) return;
    int j = (int)(t % KP);
    long long rest = t / KP;
    int l = (int)(rest % L1OUT);
    long long b = rest / L1OUT;
    _Float16 v = (_Float16)0.0f;
    if (j < C * 3) {
        int c = j / 3, dk = j - c * 3;
        v = src[((size_t)b * LSEQ + l + dk) * srcStride + c];
    }
    out[t] = v;
}

// VALID max-pool over positions, fused with f16 repack (K-pad zeros).
__global__ void pool_max_f16(const float* __restrict__ in, _Float16* __restrict__ out,
                             int inL, int outL, int win, int stride,
                             int inStride, int valid, int outStride, long long total)
{
    long long t = (long long)blockIdx.x * blockDim.x + threadIdx.x;
    if (t >= total) return;
    int c = (int)(t % outStride);
    long long rest = t / outStride;
    int p = (int)(rest % outL);
    long long b = rest / outL;
    _Float16 o = (_Float16)0.0f;
    if (c < valid) {
        float m = -3.4e38f;
        for (int w = 0; w < win; ++w)
            m = fmaxf(m, in[((size_t)b * inL + p * stride + w) * inStride + c]);
        o = (_Float16)m;
    }
    out[t] = o;
}

// before_avg = Y*Z elementwise, mean over 99 positions -> avg16 [256,256]
__global__ void mulmean(const float* __restrict__ Y, const float* __restrict__ Z,
                        _Float16* __restrict__ avg16)
{
    int t = blockIdx.x * blockDim.x + threadIdx.x;
    if (t >= NB * 256) return;
    int j = t & 255;
    int b = t >> 8;
    float s = 0.f;
    for (int l = 0; l < P2OUT; ++l) {
        size_t idx = ((size_t)b * P2OUT + l) * 256 + j;
        s += Y[idx] * Z[idx];
    }
    avg16[t] = (_Float16)(s * (1.0f / 99.0f));
}

// d_out = [logits 256x2, hf 256x128]
__global__ void write_out(const float* __restrict__ logits, const float* __restrict__ hf,
                          float* __restrict__ out)
{
    int t = blockIdx.x * blockDim.x + threadIdx.x;
    if (t < NB * 2) {
        int b = t >> 1, c = t & 1;
        out[t] = logits[b * 16 + c];
    }
    if (t < NB * 128) out[NB * 2 + t] = hf[t];
}

// ---------------------------------------------------------------------------
extern "C" void kernel_launch(void* const* d_in, const int* in_sizes, int n_in,
                              void* d_out, int out_size, void* d_ws, size_t ws_size,
                              hipStream_t stream)
{
    (void)in_sizes; (void)n_in; (void)out_size; (void)ws_size;

    const float* x    = (const float*)d_in[0];
    const int*   eidx = (const int*)d_in[1];
    const int*   etyp = (const int*)d_in[2];
    const float* ggnnW = (const float*)d_in[3];
    const float* ggnnb = (const float*)d_in[4];
    const float* Wih  = (const float*)d_in[5];
    const float* Whh  = (const float*)d_in[6];
    const float* bih  = (const float*)d_in[7];
    const float* bhh  = (const float*)d_in[8];
    const float* c1w  = (const float*)d_in[9];
    const float* c1b  = (const float*)d_in[10];
    const float* c2w  = (const float*)d_in[11];
    const float* c2b  = (const float*)d_in[12];
    const float* cc1w = (const float*)d_in[13];
    const float* cc1b = (const float*)d_in[14];
    const float* cc2w = (const float*)d_in[15];
    const float* cc2b = (const float*)d_in[16];
    const float* myw  = (const float*)d_in[17];
    const float* myb  = (const float*)d_in[18];
    const float* mzw  = (const float*)d_in[19];
    const float* mzb  = (const float*)d_in[20];
    const float* l1w  = (const float*)d_in[21];
    const float* l1b  = (const float*)d_in[22];
    const float* f1w  = (const float*)d_in[23];
    const float* f1b  = (const float*)d_in[24];
    const float* f2w  = (const float*)d_in[25];
    const float* f2b  = (const float*)d_in[26];
    const float* clsw = (const float*)d_in[27];
    const float* clsb = (const float*)d_in[28];
    const int* srcv = eidx;            // edge_index[0]
    const int* dstv = eidx + N_EDGES;  // edge_index[1]
    float* outp = (float*)d_out;

    // ---- workspace carving (~1.5 GB) ----
    char* ws = (char*)d_ws;
    size_t off = 0;
    auto carve = [&](size_t bytes) -> char* {
        char* p = ws + off;
        off += (bytes + 255) & ~(size_t)255;
        return p;
    };
    float*    h32   = (float*)   carve((size_t)N_NODES * HS * 4);
    _Float16* h16   = (_Float16*)carve((size_t)N_NODES * HK * 2);
    float*    a32   = (float*)   carve((size_t)N_NODES * HS * 4);
    _Float16* a16   = (_Float16*)carve((size_t)N_NODES * HK * 2);
    float*    Whcat = (float*)   carve((size_t)N_NODES * 416 * 4);
    float*    gi    = (float*)   carve((size_t)N_NODES * G3P * 4);
    float*    gh    = (float*)   carve((size_t)N_NODES * G3P * 4);
    _Float16* ci16  = (_Float16*)carve((size_t)N_NODES * CCK * 2);
    _Float16* imb   = (_Float16*)carve((size_t)ROWS1 * 960 * 2);
    float*    conv  = (float*)   carve((size_t)ROWS1 * 320 * 4);
    _Float16* p1h   = (_Float16*)carve((size_t)ROWSP1 * HK * 2);
    _Float16* p1c   = (_Float16*)carve((size_t)ROWSP1 * CCK * 2);
    _Float16* Y216  = (_Float16*)carve((size_t)ROWS2 * HK * 2);
    _Float16* Z216  = (_Float16*)carve((size_t)ROWS2 * CCK * 2);
    float*    Ybuf  = (float*)   carve((size_t)ROWS2 * 256 * 4);
    float*    Zbuf  = (float*)   carve((size_t)ROWS2 * 256 * 4);
    _Float16* avg16 = (_Float16*)carve((size_t)NB * 256 * 2);
    float*    h1f   = (float*)   carve((size_t)NB * 128 * 4);
    _Float16* h1h   = (_Float16*)carve((size_t)NB * 128 * 2);
    float*    q1f   = (float*)   carve((size_t)NB * 64 * 4);
    _Float16* q1h   = (_Float16*)carve((size_t)NB * 64 * 2);
    float*    hff   = (float*)   carve((size_t)NB * 128 * 4);
    _Float16* hfh   = (_Float16*)carve((size_t)NB * 128 * 2);
    float*    logf  = (float*)   carve((size_t)NB * 16 * 4);
    // packed weights (f16 fragment order) + padded biases (f32)
    _Float16* BpG  = (_Float16*)carve((size_t)26 * 7  * 512 * 2);  float* bG  = (float*)carve(416 * 4);
    _Float16* BpIH = (_Float16*)carve((size_t)38 * 7  * 512 * 2);  float* bIH = (float*)carve(G3P * 4);
    _Float16* BpHH = (_Float16*)carve((size_t)38 * 7  * 512 * 2);  float* bHH = (float*)carve(G3P * 4);
    _Float16* BpC1 = (_Float16*)carve((size_t)13 * 19 * 512 * 2);  float* bC1 = (float*)carve(HS * 4);
    _Float16* BpC2 = (_Float16*)carve((size_t)13 * 7  * 512 * 2);  float* bC2 = (float*)carve(HS * 4);
    _Float16* BpQ1 = (_Float16*)carve((size_t)20 * 30 * 512 * 2);  float* bQ1 = (float*)carve(CCK * 4);
    _Float16* BpQ2 = (_Float16*)carve((size_t)20 * 10 * 512 * 2);  float* bQ2 = (float*)carve(CCK * 4);
    _Float16* BpMY = (_Float16*)carve((size_t)16 * 7  * 512 * 2);  float* bMY = (float*)carve(256 * 4);
    _Float16* BpMZ = (_Float16*)carve((size_t)16 * 10 * 512 * 2);  float* bMZ = (float*)carve(256 * 4);
    _Float16* BpL1 = (_Float16*)carve((size_t)8  * 8  * 512 * 2);  float* bL1 = (float*)carve(128 * 4);
    _Float16* BpF1 = (_Float16*)carve((size_t)4  * 4  * 512 * 2);  float* bF1 = (float*)carve(64 * 4);
    _Float16* BpF2 = (_Float16*)carve((size_t)8  * 2  * 512 * 2);  float* bF2 = (float*)carve(128 * 4);
    _Float16* BpCL = (_Float16*)carve((size_t)1  * 4  * 512 * 2);  float* bCL = (float*)carve(16 * 4);

    auto blocks = [](long long total) { return (unsigned)((total + 255) / 256); };
    auto packW = [&](const float* W, _Float16* Bp, int O, int Ieff, int KP, int nb) {
        int ntl = (O + 15) / 16;
        long long tot = (long long)ntl * (KP >> 5) * 512;
        pack_bt<<<blocks(tot), 256, 0, stream>>>(W, Bp, O, Ieff, KP, nb, ntl);
    };
    auto packB = [&](const float* b, float* bp, int O, int nb, int cnt) {
        pack_bias<<<blocks(cnt), 256, 0, stream>>>(b, bp, O, nb, cnt);
    };
    auto gemm = [&](const _Float16* A, const _Float16* Bp, const float* bias, float* C,
                    int M, int KP, int NP, int relu) {
        dim3 g((unsigned)(M / 16), (unsigned)((NP / 16 + 3) / 4));
        gemm_f16_wmma<<<g, 128, 0, stream>>>(A, Bp, bias, C, M, KP, NP, relu);
    };

    // ---- weight packing (every call: deterministic vs poisoned ws) ----
    packW(ggnnW,             BpG, HID, HID, HK, 0);
    packW(ggnnW + HID * HID, BpG, HID, HID, HK, 13);
    packB(ggnnb,       bG, HID, 0,   HS);
    packB(ggnnb + HID, bG, HID, HS,  HS);
    packW(Wih, BpIH, 3 * HID, HID, HK, 0);  packB(bih, bIH, 3 * HID, 0, G3P);
    packW(Whh, BpHH, 3 * HID, HID, HK, 0);  packB(bhh, bHH, 3 * HID, 0, G3P);
    packW(c1w,  BpC1, HID, HID * 3, 608, 0); packB(c1b,  bC1, HID, 0, HS);
    packW(c2w,  BpC2, HID, HID,     HK,  0); packB(c2b,  bC2, HID, 0, HS);
    packW(cc1w, BpQ1, CCK, CCK * 3, 960, 0); packB(cc1b, bQ1, CCK, 0, CCK);
    packW(cc2w, BpQ2, CCK, CCK,     CCK, 0); packB(cc2b, bQ2, CCK, 0, CCK);
    packW(myw,  BpMY, 256, HID, HK,  0);     packB(myb, bMY, 256, 0, 256);
    packW(mzw,  BpMZ, 256, CCK, CCK, 0);     packB(mzb, bMZ, 256, 0, 256);
    packW(l1w,  BpL1, 128, 256, 256, 0);     packB(l1b, bL1, 128, 0, 128);
    packW(f1w,  BpF1, 64, 128, 128, 0);      packB(f1b, bF1, 64, 0, 64);
    packW(f2w,  BpF2, 128, 64, 64, 0);       packB(f2b, bF2, 128, 0, 128);
    packW(clsw, BpCL, 2, 128, 128, 0);       packB(clsb, bCL, 2, 0, 16);

    // ---- h0 = zero-pad(x) ----
    init_h<<<blocks((long long)N_NODES * HK), 256, 0, stream>>>(x, h32, h16,
                                                                (long long)N_NODES * HK);

    // ---- 8 GGNN steps ----
    for (int step = 0; step < 8; ++step) {
        // Wh for both etypes in one GEMM: [N,416]
        gemm(h16, BpG, bG, Whcat, N_NODES, HK, 416, 0);
        zero_f32<<<blocks((long long)N_NODES * HS), 256, 0, stream>>>(a32,
                                                       (long long)N_NODES * HS);
        edge_scatter<<<blocks((long long)N_EDGES * HID), 256, 0, stream>>>(
            Whcat, srcv, dstv, etyp, a32, (long long)N_EDGES * HID);
        cast_pad_f16<<<blocks((long long)N_NODES * HK), 256, 0, stream>>>(
            a32, a16, HS, HID, HK, (long long)N_NODES * HK);
        gemm(a16, BpIH, bIH, gi, N_NODES, HK, G3P, 0);
        gemm(h16, BpHH, bHH, gh, N_NODES, HK, G3P, 0);
        gru_update<<<blocks((long long)N_NODES * HID), 256, 0, stream>>>(
            gi, gh, h32, h16, (long long)N_NODES * HID);
    }

    // ---- readout: h-branch ----
    im2col3<<<blocks((long long)ROWS1 * 608), 256, 0, stream>>>(
        h16, imb, HK, HID, 608, (long long)ROWS1 * 608);
    gemm(imb, BpC1, bC1, conv, ROWS1, 608, HS, 1);                 // conv1 + relu
    pool_max_f16<<<blocks((long long)ROWSP1 * HK), 256, 0, stream>>>(
        conv, p1h, L1OUT, P1OUT, 3, 2, HS, HID, HK, (long long)ROWSP1 * HK);
    gemm(p1h, BpC2, bC2, conv, ROWSP1, HK, HS, 1);                 // conv2 + relu
    pool_max_f16<<<blocks((long long)ROWS2 * HK), 256, 0, stream>>>(
        conv, Y216, P1OUT, P2OUT, 2, 2, HS, HID, HK, (long long)ROWS2 * HK);

    // ---- readout: concat-branch ----
    ci_pack<<<blocks((long long)N_NODES * CCK), 256, 0, stream>>>(
        x, h32, ci16, (long long)N_NODES * CCK);
    im2col3<<<blocks((long long)ROWS1 * 960), 256, 0, stream>>>(
        ci16, imb, CCK, CCK, 960, (long long)ROWS1 * 960);
    gemm(imb, BpQ1, bQ1, conv, ROWS1, 960, CCK, 1);                // cconv1 + relu
    pool_max_f16<<<blocks((long long)ROWSP1 * CCK), 256, 0, stream>>>(
        conv, p1c, L1OUT, P1OUT, 3, 2, CCK, CCK, CCK, (long long)ROWSP1 * CCK);
    gemm(p1c, BpQ2, bQ2, conv, ROWSP1, CCK, CCK, 1);               // cconv2 + relu
    pool_max_f16<<<blocks((long long)ROWS2 * CCK), 256, 0, stream>>>(
        conv, Z216, P1OUT, P2OUT, 2, 2, CCK, CCK, CCK, (long long)ROWS2 * CCK);

    // ---- mlp_y / mlp_z, product + mean ----
    gemm(Y216, BpMY, bMY, Ybuf, ROWS2, HK, 256, 0);
    gemm(Z216, BpMZ, bMZ, Zbuf, ROWS2, CCK, 256, 0);
    mulmean<<<blocks((long long)NB * 256), 256, 0, stream>>>(Ybuf, Zbuf, avg16);

    // ---- metric-learning head ----
    gemm(avg16, BpL1, bL1, h1f, NB, 256, 128, 1);
    cast_pad_f16<<<blocks((long long)NB * 128), 256, 0, stream>>>(
        h1f, h1h, 128, 128, 128, (long long)NB * 128);
    gemm(h1h, BpF1, bF1, q1f, NB, 128, 64, 1);
    cast_pad_f16<<<blocks((long long)NB * 64), 256, 0, stream>>>(
        q1f, q1h, 64, 64, 64, (long long)NB * 64);
    gemm(q1h, BpF2, bF2, hff, NB, 64, 128, 1);                     // hf embedding
    cast_pad_f16<<<blocks((long long)NB * 128), 256, 0, stream>>>(
        hff, hfh, 128, 128, 128, (long long)NB * 128);
    gemm(hfh, BpCL, bCL, logf, NB, 128, 16, 0);                    // logits (NP=16, 2 valid)

    write_out<<<blocks((long long)NB * 128), 256, 0, stream>>>(logf, hff, outp);
}
